// VQEmbedding_25752623907355
// MI455X (gfx1250) — matrix operand order repaired and checked by reading, not compiled, not run on hardware
//
#include <hip/hip_runtime.h>

#define NTOK  32768
#define EMB   512
#define NCODE 4096
#define LSTRIDE (EMB + 8)   // 520 bf16 row stride in LDS -> conflict-free b128 reads

typedef __bf16 bf16;
typedef __attribute__((ext_vector_type(8)))  __bf16 v8bf;
typedef __attribute__((ext_vector_type(16))) __bf16 v16bf;
typedef __attribute__((ext_vector_type(8)))  float  v8f;

union V16 { v16bf v; v8bf c[2]; };

// ---------------------------------------------------------------------------
// K1: codebook -> bf16 hi/lo split + exact fp32 row norms ||e||^2
// ---------------------------------------------------------------------------
__global__ void __launch_bounds__(256) k_prep(const float* __restrict__ cb,
                                              bf16* __restrict__ cb_hi,
                                              bf16* __restrict__ cb_lo,
                                              float* __restrict__ e_sq) {
  __shared__ float red[256];
  const int row = blockIdx.x;
  const int tid = threadIdx.x;
  float s = 0.0f;
#pragma unroll
  for (int it = 0; it < EMB / 256; ++it) {
    const int c = tid + it * 256;
    const float x = cb[(size_t)row * EMB + c];
    const bf16 h = (bf16)x;
    const bf16 l = (bf16)(x - (float)h);
    cb_hi[(size_t)row * EMB + c] = h;
    cb_lo[(size_t)row * EMB + c] = l;
    s += x * x;
  }
  red[tid] = s;
  __syncthreads();
  for (int off = 128; off > 0; off >>= 1) {
    if (tid < off) red[tid] += red[tid + off];
    __syncthreads();
  }
  if (tid == 0) e_sq[row] = red[0];
}

// ---------------------------------------------------------------------------
// K2: fused distance + argmin with 2x2 register tiling.
//   dist = ||e||^2 - 2*z.e   (||z||^2 is row-constant -> dropped for argmin)
//   z.e ~= z_hi.e_hi + z_hi.e_lo + z_lo.e_hi  (split-bf16, ~fp32 accuracy)
// Block = 8 waves, 128 z-rows staged in LDS (bf16 hi/lo).  Wave (rg,nh):
// rows rg*32..rg*32+31 (2 M-tiles), codeword half nh (2048 cols, 2 N-tiles
// per step).  Each A/B fragment feeds 2 WMMA triples -> 2x the arithmetic
// intensity on both the LDS-A and global-B operand paths.
// ---------------------------------------------------------------------------
__global__ void __launch_bounds__(256) k_argmin(const float* __restrict__ z,
                                                const bf16* __restrict__ cb_hi,
                                                const bf16* __restrict__ cb_lo,
                                                const float* __restrict__ e_sq,
                                                int*   __restrict__ out_idx,
                                                float* __restrict__ out_idx_f) {
  extern __shared__ char smem[];
  bf16* zhi = (bf16*)smem;
  bf16* zlo = zhi + 128 * LSTRIDE;
  __shared__ float mergev[2][128];
  __shared__ int   mergei[2][128];

  const int tid = threadIdx.x;
  const int rowbase = blockIdx.x * 128;

  // Cooperative load + hi/lo convert of the 128x512 fp32 z tile.
  for (int i = tid; i < 128 * (EMB / 4); i += 256) {
    const int r = i / (EMB / 4);
    const int c = (i % (EMB / 4)) * 4;
    const float4 x4 = *(const float4*)(z + (size_t)(rowbase + r) * EMB + c);
    const float xs[4] = {x4.x, x4.y, x4.z, x4.w};
#pragma unroll
    for (int j = 0; j < 4; ++j) {
      const bf16 h = (bf16)xs[j];
      const bf16 l = (bf16)(xs[j] - (float)h);
      zhi[r * LSTRIDE + c + j] = h;
      zlo[r * LSTRIDE + c + j] = l;
    }
  }
  __syncthreads();

  const int lane = tid & 31;
  const int wave = tid >> 5;
  const int half = lane >> 4;   // 0: lanes 0-15, 1: lanes 16-31
  const int l15  = lane & 15;
  const int rg   = wave >> 1;   // row group: 32 rows
  const int nh   = wave & 1;    // codeword half: 2048 cols

  // 16-bit A operand layout (ISA 7.12.2): lane<16 holds M=lane, K{0..7,16..23};
  // lane>=16 holds M=lane-16, K{8..15,24..31}.  Chunks at +d and +d+16.
  const bf16* a_hi_base[2];
  const bf16* a_lo_base[2];
#pragma unroll
  for (int mi = 0; mi < 2; ++mi) {
    const int arow = rg * 32 + mi * 16 + l15;
    a_hi_base[mi] = zhi + arow * LSTRIDE + half * 8;
    a_lo_base[mi] = zlo + arow * LSTRIDE + half * 8;
  }

  float best[2][8];
  int   bidx[2][8];
#pragma unroll
  for (int mi = 0; mi < 2; ++mi)
#pragma unroll
    for (int j = 0; j < 8; ++j) { best[mi][j] = 3.0e38f; bidx[mi][j] = 0; }

  const int nt_begin = nh * (NCODE / 2);
  const int nt_end   = nt_begin + (NCODE / 2);

  for (int nt = nt_begin; nt < nt_end; nt += 32) {
    // B operand 32x16: lane<16 holds col, K{0..7,16..23}; lane>=16 K{8..15,24..31}.
    const bf16* b_hi_base[2];
    const bf16* b_lo_base[2];
#pragma unroll
    for (int ni = 0; ni < 2; ++ni) {
      const int ncol = nt + ni * 16 + l15;
      b_hi_base[ni] = cb_hi + (size_t)ncol * EMB + half * 8;
      b_lo_base[ni] = cb_lo + (size_t)ncol * EMB + half * 8;
    }

    v8f acc[2][2] = {};
#pragma unroll 2
    for (int d = 0; d < EMB; d += 32) {
      V16 a_h[2], a_l[2], b_h[2], b_l[2];
#pragma unroll
      for (int mi = 0; mi < 2; ++mi) {
        a_h[mi].c[0] = *(const v8bf*)(a_hi_base[mi] + d);
        a_h[mi].c[1] = *(const v8bf*)(a_hi_base[mi] + d + 16);
        a_l[mi].c[0] = *(const v8bf*)(a_lo_base[mi] + d);
        a_l[mi].c[1] = *(const v8bf*)(a_lo_base[mi] + d + 16);
      }
#pragma unroll
      for (int ni = 0; ni < 2; ++ni) {
        b_h[ni].c[0] = *(const v8bf*)(b_hi_base[ni] + d);
        b_h[ni].c[1] = *(const v8bf*)(b_hi_base[ni] + d + 16);
        b_l[ni].c[0] = *(const v8bf*)(b_lo_base[ni] + d);
        b_l[ni].c[1] = *(const v8bf*)(b_lo_base[ni] + d + 16);
      }
#pragma unroll
      for (int mi = 0; mi < 2; ++mi)
#pragma unroll
        for (int ni = 0; ni < 2; ++ni) {
          acc[mi][ni] = __builtin_amdgcn_wmma_f32_16x16x32_bf16(
              false, a_h[mi].v, false, b_h[ni].v, (short)0, acc[mi][ni], false, false);
          acc[mi][ni] = __builtin_amdgcn_wmma_f32_16x16x32_bf16(
              false, a_h[mi].v, false, b_l[ni].v, (short)0, acc[mi][ni], false, false);
          acc[mi][ni] = __builtin_amdgcn_wmma_f32_16x16x32_bf16(
              false, a_l[mi].v, false, b_h[ni].v, (short)0, acc[mi][ni], false, false);
        }
    }

    // C/D layout: lane holds one column; elem j = row (half*8 + j).
#pragma unroll
    for (int ni = 0; ni < 2; ++ni) {
      const int ncol = nt + ni * 16 + l15;
      const float es = e_sq[ncol];
#pragma unroll
      for (int mi = 0; mi < 2; ++mi)
#pragma unroll
        for (int j = 0; j < 8; ++j) {
          const float dist = es - 2.0f * acc[mi][ni][j];
          if (dist < best[mi][j]) { best[mi][j] = dist; bidx[mi][j] = ncol; }
        }
    }

    if (nt + 32 < nt_end) {  // pull next codebook tile toward the WGP
      __builtin_prefetch(cb_hi + (size_t)(nt + 32 + l15) * EMB, 0, 1);
      __builtin_prefetch(cb_lo + (size_t)(nt + 32 + l15) * EMB, 0, 1);
    }
  }

  // Cross-lane argmin: row (half*8+j) lives in all 16 lanes of this half.
#pragma unroll
  for (int m = 1; m < 16; m <<= 1) {
#pragma unroll
    for (int mi = 0; mi < 2; ++mi)
#pragma unroll
      for (int j = 0; j < 8; ++j) {
        const float ov = __shfl_xor(best[mi][j], m, 16);
        const int   oi = __shfl_xor(bidx[mi][j], m, 16);
        if (ov < best[mi][j] || (ov == best[mi][j] && oi < bidx[mi][j])) {
          best[mi][j] = ov; bidx[mi][j] = oi;
        }
      }
  }
  // Cross-wave merge: the two column-halves of each row meet in LDS.
  if (l15 == 0) {
#pragma unroll
    for (int mi = 0; mi < 2; ++mi)
#pragma unroll
      for (int j = 0; j < 8; ++j) {
        const int rloc = rg * 32 + mi * 16 + half * 8 + j;
        mergev[nh][rloc] = best[mi][j];
        mergei[nh][rloc] = bidx[mi][j];
      }
  }
  __syncthreads();
  if (tid < 128) {
    float v0 = mergev[0][tid];
    int   i0 = mergei[0][tid];
    const float v1 = mergev[1][tid];
    const int   i1 = mergei[1][tid];
    if (v1 < v0 || (v1 == v0 && i1 < i0)) { v0 = v1; i0 = i1; }
    const int row = rowbase + tid;
    out_idx[row]   = i0;
    out_idx_f[row] = (float)i0;
  }
}

// ---------------------------------------------------------------------------
// K3: exact fp32 gather z_q = codebook[idx] + double-precision loss partials.
// Forward values: z_q_st == z_q ; loss = 1.25 * mean((z_q - z)^2).
// ---------------------------------------------------------------------------
__global__ void __launch_bounds__(256) k_gather_loss(const float* __restrict__ z,
                                                     const float* __restrict__ cb,
                                                     const int*   __restrict__ idx,
                                                     float* __restrict__ zq_out,
                                                     double* __restrict__ acc) {
  __shared__ double red[256];
  const int row = blockIdx.x;
  const int tid = threadIdx.x;
  const int k = idx[row];
  double s = 0.0;
#pragma unroll
  for (int it = 0; it < EMB / 256; ++it) {
    const int c = tid + it * 256;
    const float q = cb[(size_t)k * EMB + c];
    const float x = z[(size_t)row * EMB + c];
    zq_out[(size_t)row * EMB + c] = q;
    const float d = q - x;
    s += (double)d * (double)d;
  }
  red[tid] = s;
  __syncthreads();
  for (int off = 128; off > 0; off >>= 1) {
    if (tid < off) red[tid] += red[tid + off];
    __syncthreads();
  }
  if (tid == 0) atomicAdd(acc, red[0]);
}

__global__ void k_final(const double* __restrict__ acc, float* __restrict__ loss) {
  *loss = (float)(1.25 * (*acc) / ((double)NTOK * (double)EMB));
}

// ---------------------------------------------------------------------------
extern "C" void kernel_launch(void* const* d_in, const int* in_sizes, int n_in,
                              void* d_out, int out_size, void* d_ws, size_t ws_size,
                              hipStream_t stream) {
  const float* z  = (const float*)d_in[0];
  const float* cb = (const float*)d_in[1];
  float* out = (float*)d_out;
  char*  ws  = (char*)d_ws;

  // Workspace layout (~8.2 MB)
  bf16*  cb_hi = (bf16*)ws;                                        // 4 MB
  bf16*  cb_lo = (bf16*)(ws + (size_t)NCODE * EMB * sizeof(bf16)); // 4 MB
  float* e_sq  = (float*)(ws + (size_t)NCODE * EMB * sizeof(bf16) * 2);
  int*   idx   = (int*)((char*)e_sq + NCODE * sizeof(float));
  double* acc  = (double*)((char*)idx + NTOK * sizeof(int));

  // d_out tuple layout: [z_q_st : N*D][loss : 1][indices(as float) : N]
  float* zq_out = out;
  float* loss   = out + (size_t)NTOK * EMB;
  float* idx_f  = loss + 1;

  hipMemsetAsync(acc, 0, sizeof(double), stream);

  k_prep<<<NCODE, 256, 0, stream>>>(cb, cb_hi, cb_lo, e_sq);

  const size_t lds_bytes = (size_t)128 * LSTRIDE * sizeof(bf16) * 2;  // 260 KB dynamic
  k_argmin<<<NTOK / 128, 256, lds_bytes, stream>>>(z, cb_hi, cb_lo, e_sq, idx, idx_f);

  k_gather_loss<<<NTOK, 256, 0, stream>>>(z, cb, idx, zq_out, acc);
  k_final<<<1, 1, 0, stream>>>(acc, loss);
}
